// LSTM_80805514707461
// MI455X (gfx1250) — matrix operand order repaired and checked by reading, not compiled, not run on hardware
//
#include <hip/hip_runtime.h>

// ---------------------------------------------------------------------------
// Fused 2-layer LSTM (B=128, T=1024, I=128, H=256) + FC head for gfx1250.
//
// Strategy (MI455X):
//  * Latency-bound recurrence -> one persistent kernel, 8 WGs (batch tile 16),
//    512 thr = 16 waves. h/c state lives on-chip for all 1024 steps.
//  * Gates via v_wmma_f32_16x16x32_f16: A = [x_t | h] rows staged in LDS (f16),
//    B = weights pre-swizzled to WMMA fragment order in d_ws (f16, ~1.8MB,
//    permanently L2-resident at 192MB L2), streamed as coalesced
//    global_load_b128 every step.
//  * Software-pipelined B stream, enforced with sched_barrier(0): the machine
//    scheduler otherwise sinks prefetches back to their consumers (observed in
//    rev 3/4 asm: every WMMA behind s_wait_loadcnt 0x0 on one reg pair).
//    The fence keeps ks+1's 8 loads issued before ks's 4 WMMAs -> ~8 loads in
//    flight per wave (x4 waves/SIMD) covering L2 latency.
//  * A laundered zero *offset* defeats LICM hoisting of the loop-invariant
//    weight loads (rev-1 spill bug) while preserving pointer provenance so
//    loads stay global_load (not flat_load coupling LOADcnt with DScnt).
//  * Wave w owns gate cols [16w,16w+16) of i,f,g,o -> element-wise LSTM cell
//    update fully in registers; c state never leaves VGPRs. Native tanh/exp.
// ---------------------------------------------------------------------------

typedef __attribute__((ext_vector_type(16))) _Float16 v16h;
typedef __attribute__((ext_vector_type(8)))  _Float16 v8h;
typedef __attribute__((ext_vector_type(8)))  float    v8f;

#define SEQ_T 1024
#define ROW0  392   // halves per A0 row: K=384 (+8 pad vs 64-bank conflicts)
#define ROW1  520   // halves per A1 row: K=512 (+8 pad)
#define KS0   12    // K-steps layer0: 384/32
#define KS1   16    // K-steps layer1: 512/32

// ---- workspace layout (bytes) ----
#define WS_W0   0                               // 64*12*512 halves = 786432 B
#define WS_W1   (786432)                        // 64*16*512 halves = 1048576 B
#define WS_B0   (786432 + 1048576)              // 1024 f32
#define WS_B1   (786432 + 1048576 + 4096)       // 1024 f32

#if __has_builtin(__builtin_amdgcn_sched_barrier)
#define SCHED_FENCE() __builtin_amdgcn_sched_barrier(0)
#else
#define SCHED_FENCE()
#endif

__device__ __forceinline__ float fast_sigmoid(float x) {
  return 1.0f / (1.0f + __expf(-x));
}

#if __has_builtin(__builtin_amdgcn_tanh_f32)
__device__ __forceinline__ float fast_tanh(float x) {
  return __builtin_amdgcn_tanh_f32(x);   // native v_tanh_f32
}
#else
__device__ __forceinline__ float fast_tanh(float x) {
  // tanh(x) = 1 - 2/(exp(2x)+1); saturates correctly at +/-1 (exp->inf/0)
  float e = __expf(2.0f * x);            // native v_exp_f32
  return 1.0f - 2.0f / (e + 1.0f);
}
#endif

// Optimization barrier on a zero index: redefined every loop iteration so
// LICM cannot hoist loads addressed with it, but the base pointer's address
// space provenance (global) is preserved.
__device__ __forceinline__ unsigned opaque_zero() {
  unsigned z = 0;
  asm volatile("" : "+s"(z));
  return z;
}

// Pack [Wxh | Whh] (f32, row = gate output n, col = k) into f16 WMMA
// B-fragment order. Linear half layout matches the main kernel's loads:
//   halfIdx = blk*512 + half16*256 + lane*8 + j,  blk = nt*KSTEPS + ks
//   (n = nt*16 + lane%16,  k = ks*32 + (lane>=16?16:0) + half16*8 + j)
__global__ void pack_weights_kernel(const float* __restrict__ Wx,
                                    const float* __restrict__ Wh,
                                    _Float16* __restrict__ dst,
                                    int KX, int KH, int KSTEPS) {
  int idx = blockIdx.x * blockDim.x + threadIdx.x;
  int total = 64 * KSTEPS * 512;
  if (idx >= total) return;
  int blk    = idx >> 9;
  int rem    = idx & 511;
  int half16 = rem >> 8;
  int rem2   = rem & 255;
  int lane   = rem2 >> 3;
  int j      = rem2 & 7;
  int jf     = half16 * 8 + j;
  int nt = blk / KSTEPS;
  int ks = blk % KSTEPS;
  int n  = nt * 16 + (lane & 15);
  int k  = ks * 32 + (lane >> 4) * 16 + jf;
  float v = (k < KX) ? Wx[n * KX + k] : Wh[n * KH + (k - KX)];
  dst[idx] = (_Float16)v;
}

__global__ void pack_bias_kernel(const float* __restrict__ bx0,
                                 const float* __restrict__ bh0,
                                 const float* __restrict__ bx1,
                                 const float* __restrict__ bh1,
                                 float* __restrict__ b0,
                                 float* __restrict__ b1) {
  int i = blockIdx.x * blockDim.x + threadIdx.x;
  if (i < 1024)       b0[i] = bx0[i] + bh0[i];
  else if (i < 2048)  b1[i - 1024] = bx1[i - 1024] + bh1[i - 1024];
}

// One K-sweep of the gate GEMM for this wave's 4 N-tiles (i,f,g,o cols).
// B fragments double-buffered one K-step ahead; sched fences pin the
// prefetch loads above the current K-step's WMMAs.
template <int KSTEPS, int ROWB>
__device__ __forceinline__ void layer_mm(const _Float16* __restrict__ Abase,
                                         const _Float16* __restrict__ W,
                                         unsigned zoff,
                                         int wave, int lane, v8f (&acc)[4]) {
  const int row = lane & 15;
  const int hi  = lane >> 4;
  const _Float16* Wl = W + zoff + (lane << 3);

  auto loadB = [&](int ks, int q, v16h& dst) {
    const _Float16* pw =
        Wl + ((size_t)((wave + 16 * q) * KSTEPS + ks) << 9);
    v8h blo = *(const v8h*)pw;          // global_load_b128 (coalesced, L2)
    v8h bhi = *(const v8h*)(pw + 256);  // global_load_b128 (coalesced, L2)
    #pragma unroll
    for (int j = 0; j < 8; ++j) { dst[j] = blo[j]; dst[8 + j] = bhi[j]; }
  };

  v16h bb[2][4];                        // double-buffered B fragments
  #pragma unroll
  for (int q = 0; q < 4; ++q) loadB(0, q, bb[0][q]);

  #pragma unroll
  for (int ks = 0; ks < KSTEPS; ++ks) {
    const int cur = ks & 1;
    if (ks + 1 < KSTEPS) {              // prefetch next K-step's 4 fragments
      #pragma unroll
      for (int q = 0; q < 4; ++q) loadB(ks + 1, q, bb[cur ^ 1][q]);
    }
    SCHED_FENCE();  // forbid sinking the prefetch below this K-step's WMMAs
    // A fragment (16x32 f16): lane<16 -> K {0..15}, lane>=16 -> K {16..31}
    const _Float16* pa = Abase + row * ROWB + ks * 32 + hi * 16;
    v8h alo = *(const v8h*)pa;          // ds_load_b128
    v8h ahi = *(const v8h*)(pa + 8);    // ds_load_b128
    v16h a;
    #pragma unroll
    for (int j = 0; j < 8; ++j) { a[j] = alo[j]; a[8 + j] = ahi[j]; }
    #pragma unroll
    for (int q = 0; q < 4; ++q)
      acc[q] = __builtin_amdgcn_wmma_f32_16x16x32_f16(
          false, a, false, bb[cur][q], (short)0, acc[q], false, false);
    SCHED_FENCE();  // keep the pipeline stage boundaries intact
  }
}

__global__ __launch_bounds__(512, 1)
void lstm2_fused_kernel(const float* __restrict__ x,
                        const _Float16* __restrict__ w0,
                        const _Float16* __restrict__ w1,
                        const float* __restrict__ b0,
                        const float* __restrict__ b1,
                        const float* __restrict__ Wfc,
                        const float* __restrict__ bfc,
                        float* __restrict__ out) {
  __shared__ __align__(16) _Float16 A0[16 * ROW0]; // [x_t | h0]  K=384
  __shared__ __align__(16) _Float16 A1[16 * ROW1]; // [h0_t | h1] K=512

  const int tid  = threadIdx.x;
  const int lane = tid & 31;
  const int wave = tid >> 5;
  const int col  = lane & 15;
  const int hi   = lane >> 4;
  const int wg   = blockIdx.x;

  for (int i = tid; i < 16 * ROW0; i += 512) A0[i] = (_Float16)0.f;
  for (int i = tid; i < 16 * ROW1; i += 512) A1[i] = (_Float16)0.f;

  const v8f zero8 = {0.f, 0.f, 0.f, 0.f, 0.f, 0.f, 0.f, 0.f};
  v8f c0 = zero8, c1 = zero8;  // cell state tiles live in VGPRs

  // Fused biases for this wave's gate columns (constant over t).
  const int nb = wave * 16 + col;
  const float bi0 = b0[nb], bf0 = b0[256 + nb], bg0 = b0[512 + nb], bo0 = b0[768 + nb];
  const float bi1 = b1[nb], bf1 = b1[256 + nb], bg1 = b1[512 + nb], bo1 = b1[768 + nb];

  __syncthreads();

  #pragma unroll 1
  for (int t = 0; t < SEQ_T; ++t) {
    // ---- stage x_t (f32->f16) into A0 x-region: 16 rows x 128 cols ----
    {
      int idx = tid << 2;            // 512 thr * 4 floats = 2048
      int r   = idx >> 7;
      int ii  = idx & 127;
      const float4 v =
          *(const float4*)(x + (((size_t)(wg * 16 + r)) * SEQ_T + t) * 128 + ii);
      _Float16* pd = &A0[r * ROW0 + ii];
      pd[0] = (_Float16)v.x; pd[1] = (_Float16)v.y;
      pd[2] = (_Float16)v.z; pd[3] = (_Float16)v.w;
    }
    __syncthreads();

    // ---- layer 0 ----
    {
      v8f acc[4] = {zero8, zero8, zero8, zero8};
      layer_mm<KS0, ROW0>(A0, w0, opaque_zero(), wave, lane, acc);
      _Float16 hbuf[8];
      #pragma unroll
      for (int r = 0; r < 8; ++r) {
        float iv = fast_sigmoid(acc[0][r] + bi0);
        float fv = fast_sigmoid(acc[1][r] + bf0);
        float gv = fast_tanh(acc[2][r] + bg0);
        float ov = fast_sigmoid(acc[3][r] + bo0);
        float cv = fv * c0[r] + iv * gv;
        c0[r] = cv;
        hbuf[r] = (_Float16)(ov * fast_tanh(cv));
      }
      __syncthreads();  // all waves finished reading A0/A1 h-regions
      #pragma unroll
      for (int r = 0; r < 8; ++r) {
        int m = hi * 8 + r;
        _Float16 hv = hbuf[r];
        A0[m * ROW0 + 128 + wave * 16 + col] = hv;  // next step's layer-0 A
        A1[m * ROW1 +       wave * 16 + col] = hv;  // this step's layer-1 A
      }
      __syncthreads();
    }

    // ---- layer 1 ----
    {
      v8f acc[4] = {zero8, zero8, zero8, zero8};
      layer_mm<KS1, ROW1>(A1, w1, opaque_zero(), wave, lane, acc);
      _Float16 hbuf[8];
      #pragma unroll
      for (int r = 0; r < 8; ++r) {
        float iv = fast_sigmoid(acc[0][r] + bi1);
        float fv = fast_sigmoid(acc[1][r] + bf1);
        float gv = fast_tanh(acc[2][r] + bg1);
        float ov = fast_sigmoid(acc[3][r] + bo1);
        float cv = fv * c1[r] + iv * gv;
        c1[r] = cv;
        hbuf[r] = (_Float16)(ov * fast_tanh(cv));
      }
      __syncthreads();
      #pragma unroll
      for (int r = 0; r < 8; ++r) {
        int m = hi * 8 + r;
        A1[m * ROW1 + 256 + wave * 16 + col] = hbuf[r];
      }
      __syncthreads();
    }
  }

  // ---- FC head on final h1 (still resident in LDS) ----
  if (tid < 160) {
    int row = tid / 10, oc = tid % 10;
    float s = bfc[oc];
    const _Float16* ph = &A1[row * ROW1 + 256];
    #pragma unroll 8
    for (int kk = 0; kk < 256; ++kk)
      s += (float)ph[kk] * Wfc[oc * 256 + kk];
    out[(wg * 16 + row) * 10 + oc] = s;
  }
}

extern "C" void kernel_launch(void* const* d_in, const int* in_sizes, int n_in,
                              void* d_out, int out_size, void* d_ws, size_t ws_size,
                              hipStream_t stream) {
  (void)in_sizes; (void)n_in; (void)out_size; (void)ws_size;
  const float* x    = (const float*)d_in[0];
  const float* Wxh0 = (const float*)d_in[1];
  const float* bxh0 = (const float*)d_in[2];
  const float* Whh0 = (const float*)d_in[3];
  const float* bhh0 = (const float*)d_in[4];
  const float* Wxh1 = (const float*)d_in[5];
  const float* bxh1 = (const float*)d_in[6];
  const float* Whh1 = (const float*)d_in[7];
  const float* bhh1 = (const float*)d_in[8];
  const float* Wfc  = (const float*)d_in[9];
  const float* bfc  = (const float*)d_in[10];

  char* ws = (char*)d_ws;
  _Float16* w0 = (_Float16*)(ws + WS_W0);
  _Float16* w1 = (_Float16*)(ws + WS_W1);
  float*    b0 = (float*)(ws + WS_B0);
  float*    b1 = (float*)(ws + WS_B1);

  // prep: pack weights to f16 WMMA fragment order + fuse biases (~1.8MB)
  pack_weights_kernel<<<(64 * KS0 * 512 + 255) / 256, 256, 0, stream>>>(
      Wxh0, Whh0, w0, 128, 256, KS0);
  pack_weights_kernel<<<(64 * KS1 * 512 + 255) / 256, 256, 0, stream>>>(
      Wxh1, Whh1, w1, 256, 256, KS1);
  pack_bias_kernel<<<8, 256, 0, stream>>>(bxh0, bhh0, bxh1, bhh1, b0, b1);

  // fused persistent LSTM: 8 batch tiles x 512 threads (16 waves)
  lstm2_fused_kernel<<<8, 512, 0, stream>>>(x, w0, w1, b0, b1, Wfc, bfc,
                                            (float*)d_out);
}
